// TitansLMModel_34127810134609
// MI455X (gfx1250) — compile-verified, matrix operand order checked by reading, not run on hardware
//
#include <hip/hip_runtime.h>

// ---------------------------------------------------------------------------
// Model dimensions (TitansLM reference)
// ---------------------------------------------------------------------------
constexpr int kVocab = 32000;
constexpr int kD     = 1024;
constexpr int kL     = 4;
constexpr int kT     = 2048;
constexpr int kM     = 64;
constexpr int kH     = 16;
constexpr int kHS    = 64;          // kD / kH
constexpr int kF     = 4096;        // 4 * kD
constexpr int kS     = kM + kT;     // 2112
constexpr int kSP    = 2176;        // kS padded to 17*128 (guard-free GEMM)

// ---------------------------------------------------------------------------
// WMMA vector types (gfx1250, wave32)
// ---------------------------------------------------------------------------
typedef __attribute__((ext_vector_type(16))) __bf16 v16bf;
typedef __attribute__((ext_vector_type(8)))  __bf16 v8bf;
typedef __attribute__((ext_vector_type(4)))  __bf16 v4bf;
typedef __attribute__((ext_vector_type(8)))  float  v8f;

static __device__ __forceinline__ v8f zero8() {
  v8f z;
#pragma unroll
  for (int i = 0; i < 8; ++i) z[i] = 0.f;
  return z;
}

// Load a 16-half WMMA A/B fragment from LDS (K-contiguous tile rows).
// CDNA5 16-bit layout: lane l holds row (l&15); kbase = (l>>4)*8;
//   frag elems 0..7  -> K = kbase + 0..7
//   frag elems 8..15 -> K = kbase + 16..23
// => exactly two ds_load_b128 per fragment.
static __device__ __forceinline__ v16bf load_frag16(const __bf16* p) {
  v8bf lo = *(const v8bf*)(p);
  v8bf hi = *(const v8bf*)(p + 16);
  return __builtin_shufflevector(lo, hi, 0, 1, 2, 3, 4, 5, 6, 7,
                                         8, 9, 10, 11, 12, 13, 14, 15);
}

// fp32x4 -> packed bf16x4, one 8-byte LDS store
static __device__ __forceinline__ void store4bf(__bf16* p, float4 v) {
  v4bf w;
  w[0] = (__bf16)v.x; w[1] = (__bf16)v.y; w[2] = (__bf16)v.z; w[3] = (__bf16)v.w;
  *(v4bf*)p = w;
}

// ---------------------------------------------------------------------------
// Embedding: x[t, :] = tok_emb[idx[t], :] + pos_emb[t, :]
// ---------------------------------------------------------------------------
__global__ __launch_bounds__(256)
void embed_kernel(const int* __restrict__ idx, const float* __restrict__ tok,
                  const float* __restrict__ pos, float* __restrict__ x) {
  const int t   = blockIdx.x;
  const int tid = threadIdx.x;
  const size_t toff = (size_t)idx[t] * kD;
#pragma unroll
  for (int i = 0; i < 4; ++i) {
    const int c = tid + i * 256;
    x[(size_t)t * kD + c] = tok[toff + c] + pos[(size_t)t * kD + c];
  }
}

// ---------------------------------------------------------------------------
// xa = concat(mem, x) along sequence
// ---------------------------------------------------------------------------
__global__ __launch_bounds__(256)
void build_xa_kernel(const float* __restrict__ mem, const float* __restrict__ x,
                     float* __restrict__ xa) {
  const int r   = blockIdx.x;
  const int tid = threadIdx.x;
  const float* src = (r < kM) ? (mem + (size_t)r * kD)
                              : (x + (size_t)(r - kM) * kD);
#pragma unroll
  for (int i = 0; i < 4; ++i) {
    const int c = tid + i * 256;
    xa[(size_t)r * kD + c] = src[c];
  }
}

// ---------------------------------------------------------------------------
// LayerNorm over last dim (1024), one block per row
// ---------------------------------------------------------------------------
__global__ __launch_bounds__(256)
void layernorm_kernel(const float* __restrict__ in, float* __restrict__ out,
                      const float* __restrict__ g, const float* __restrict__ b) {
  __shared__ float sred[8];
  __shared__ float sbc[2];
  const int row = blockIdx.x;
  const int tid = threadIdx.x;
  const float* xr = in + (size_t)row * kD;
  const float v0 = xr[tid], v1 = xr[tid + 256], v2 = xr[tid + 512], v3 = xr[tid + 768];

  float s = v0 + v1 + v2 + v3;
#pragma unroll
  for (int off = 16; off > 0; off >>= 1) s += __shfl_xor(s, off);
  if ((tid & 31) == 0) sred[tid >> 5] = s;
  __syncthreads();
  if (tid == 0) {
    float r = 0.f;
    for (int i = 0; i < 8; ++i) r += sred[i];
    sbc[0] = r;
  }
  __syncthreads();
  const float mean = sbc[0] * (1.0f / kD);
  const float d0 = v0 - mean, d1 = v1 - mean, d2 = v2 - mean, d3 = v3 - mean;

  float q = d0 * d0 + d1 * d1 + d2 * d2 + d3 * d3;
#pragma unroll
  for (int off = 16; off > 0; off >>= 1) q += __shfl_xor(q, off);
  __syncthreads();
  if ((tid & 31) == 0) sred[tid >> 5] = q;
  __syncthreads();
  if (tid == 0) {
    float r = 0.f;
    for (int i = 0; i < 8; ++i) r += sred[i];
    sbc[1] = r;
  }
  __syncthreads();
  const float rstd = rsqrtf(sbc[1] * (1.0f / kD) + 1e-5f);

  float* orow = out + (size_t)row * kD;
  orow[tid]       = d0 * rstd * g[tid]       + b[tid];
  orow[tid + 256] = d1 * rstd * g[tid + 256] + b[tid + 256];
  orow[tid + 512] = d2 * rstd * g[tid + 512] + b[tid + 512];
  orow[tid + 768] = d3 * rstd * g[tid + 768] + b[tid + 768];
}

// ---------------------------------------------------------------------------
// Repack per-head projection weight: W[h, d, e] -> out[d, h*HS + e]
// ---------------------------------------------------------------------------
__global__ __launch_bounds__(256)
void pack_qkv_kernel(const float* __restrict__ W, float* __restrict__ out) {
  const int idx = blockIdx.x * 256 + threadIdx.x;   // 0 .. D*D-1
  const int d = idx >> 10;
  const int n = idx & 1023;
  const int h = n >> 6;
  const int e = n & 63;
  out[idx] = W[((size_t)h << 16) + ((size_t)d << 6) + e];   // h*D*HS + d*HS + e
}

// ---------------------------------------------------------------------------
// Guard-free bf16-WMMA GEMM:  C[M,N] = A[M,K]*B[K,N] (+bias)(+resid)(relu?)
// REQUIRES: Md % 128 == 0, Nd % 128 == 0, Kd % 64 == 0 (caller pads M).
// Block tile 128x128x64; 8 waves (2x4), each wave 64x32 = 8 accumulators.
// Software-pipelined: next panel's 16 global_load_b128 are all in flight
// while the current panel's 16 WMMAs execute.
// ---------------------------------------------------------------------------
constexpr int BM = 128, BN = 128, BK = 64;
constexpr int LDAS = BK + 8;   // 72 halves -> 144B rows (16B aligned)
constexpr int LDBS = BK + 8;

static __device__ __forceinline__ void panel_load(
    const float* __restrict__ A, const float* __restrict__ B,
    int bm0, int bn0, int k0, int Kd, int Nd, int tid,
    float4* ta, float4* tb) {
#pragma unroll
  for (int i = 0; i < 8; ++i) {
    const int idx4 = (i * 256 + tid) * 4;
    ta[i] = *(const float4*)(A + (size_t)(bm0 + (idx4 >> 6)) * Kd + k0 + (idx4 & 63));
  }
#pragma unroll
  for (int i = 0; i < 8; ++i) {
    const int idx4 = (i * 256 + tid) * 4;
    tb[i] = *(const float4*)(B + (size_t)(k0 + (idx4 >> 7)) * Nd + bn0 + (idx4 & 127));
  }
}

__global__ __launch_bounds__(256)
void gemm_bf16_kernel(const float* __restrict__ A, const float* __restrict__ B,
                      float* __restrict__ C, const float* __restrict__ bias,
                      const float* __restrict__ resid,
                      int Md, int Nd, int Kd, int relu) {
  __shared__ __align__(16) __bf16 As[BM][LDAS];     // [m][k]
  __shared__ __align__(16) __bf16 Bs[BN][LDBS];     // [n][k] (transposed)

  const int tid    = threadIdx.x;
  const int lane   = tid & 31;
  const int wave   = tid >> 5;     // 0..7
  const int wm     = wave >> 2;    // 0..1 : 64-row band
  const int wn     = wave & 3;     // 0..3 : 32-col band
  const int l15    = lane & 15;
  const int khalf  = (lane >> 4) * 8;
  const int rowoff = khalf;        // C-layout row offset for lanes 16..31

  const int bm0 = blockIdx.y * BM;
  const int bn0 = blockIdx.x * BN;

  v8f acc[4][2];
#pragma unroll
  for (int i = 0; i < 4; ++i)
#pragma unroll
    for (int j = 0; j < 2; ++j) acc[i][j] = zero8();

  float4 ta[8], tb[8];
  panel_load(A, B, bm0, bn0, 0, Kd, Nd, tid, ta, tb);   // prologue

  for (int k0 = 0; k0 < Kd; k0 += BK) {
    // ---- commit staged panel to LDS (packed bf16) ----
#pragma unroll
    for (int i = 0; i < 8; ++i) {
      const int idx4 = (i * 256 + tid) * 4;
      store4bf(&As[idx4 >> 6][idx4 & 63], ta[i]);
    }
#pragma unroll
    for (int i = 0; i < 8; ++i) {
      const int idx4 = (i * 256 + tid) * 4;
      const int r = idx4 >> 7, c = idx4 & 127;     // r = k, c = n
      Bs[c + 0][r] = (__bf16)tb[i].x;
      Bs[c + 1][r] = (__bf16)tb[i].y;
      Bs[c + 2][r] = (__bf16)tb[i].z;
      Bs[c + 3][r] = (__bf16)tb[i].w;
    }
    __syncthreads();

    // ---- kick off next panel's global loads (latency hidden by WMMAs) ----
    if (k0 + BK < Kd)
      panel_load(A, B, bm0, bn0, k0 + BK, Kd, Nd, tid, ta, tb);

    // ---- 16 WMMAs on the committed panel ----
#pragma unroll
    for (int kk = 0; kk < BK; kk += 32) {
      v16bf af[4], bfr[2];
#pragma unroll
      for (int i = 0; i < 4; ++i)
        af[i] = load_frag16(&As[wm * 64 + i * 16 + l15][kk + khalf]);
#pragma unroll
      for (int j = 0; j < 2; ++j)
        bfr[j] = load_frag16(&Bs[wn * 32 + j * 16 + l15][kk + khalf]);
#pragma unroll
      for (int i = 0; i < 4; ++i)
#pragma unroll
        for (int j = 0; j < 2; ++j)
          acc[i][j] = __builtin_amdgcn_wmma_f32_16x16x32_bf16(
              false, af[i], false, bfr[j], (short)0, acc[i][j], false, false);
    }
    __syncthreads();
  }

  // ---- epilogue: bias + residual + relu, C-layout store (no guards) ----
#pragma unroll
  for (int i = 0; i < 4; ++i)
#pragma unroll
    for (int j = 0; j < 2; ++j) {
      const int mB = bm0 + wm * 64 + i * 16 + rowoff;
      const int n  = bn0 + wn * 32 + j * 16 + l15;
      const float bia = bias ? bias[n] : 0.f;
#pragma unroll
      for (int r = 0; r < 8; ++r) {
        const int m = mB + r;
        float v = acc[i][j][r] + bia;
        if (resid) v += resid[(size_t)m * Nd + n];
        if (relu)  v = fmaxf(v, 0.f);
        C[(size_t)m * Nd + n] = v;
      }
    }
}

// ---------------------------------------------------------------------------
// Flash attention, one wave per (16-query tile, head).
// Q/K/V laid out [s][h*HS + e]. Causal mask over the full S domain.
// ---------------------------------------------------------------------------
__global__ __launch_bounds__(32)
void attn_kernel(const float* __restrict__ Q, const float* __restrict__ K,
                 const float* __restrict__ Vm, float* __restrict__ O) {
  __shared__ __align__(16) __bf16 Qs[16][72];
  __shared__ __align__(16) __bf16 Ks[32][72];
  __shared__ __align__(16) __bf16 Vt[64][40];   // [e][key] (transposed)
  __shared__ __align__(16) __bf16 Ps[16][40];   // [q][key]

  const int h      = blockIdx.y;
  const int q0     = blockIdx.x * 16;
  const int lane   = threadIdx.x;
  const int l15    = lane & 15;
  const int rowoff = (lane >> 4) * 8;
  const int khalf  = rowoff;

  // stage the 16x64 Q tile (loads batched in registers first)
  {
    float4 t[8];
#pragma unroll
    for (int i = 0; i < 8; ++i) {
      const int idx4 = (i * 32 + lane) * 4;
      t[i] = *(const float4*)(Q + (size_t)(q0 + (idx4 >> 6)) * kD + h * kHS + (idx4 & 63));
    }
#pragma unroll
    for (int i = 0; i < 8; ++i) {
      const int idx4 = (i * 32 + lane) * 4;
      store4bf(&Qs[idx4 >> 6][idx4 & 63], t[i]);
    }
  }
  __syncthreads();

  const v16bf qf0 = load_frag16(&Qs[l15][khalf]);
  const v16bf qf1 = load_frag16(&Qs[l15][32 + khalf]);

  v8f oacc[4];
#pragma unroll
  for (int t = 0; t < 4; ++t) oacc[t] = zero8();
  float rmax[8], rsum[8];
#pragma unroll
  for (int r = 0; r < 8; ++r) { rmax[r] = -3.0e38f; rsum[r] = 0.f; }

  const int   qmax  = q0 + 15;
  const float scale = 0.125f;   // HS^-0.5

  for (int kb0 = 0; kb0 <= qmax; kb0 += 32) {
    // stage K (row-major) and V (transposed), loads batched in registers
#pragma unroll
    for (int g = 0; g < 2; ++g) {
      float4 t[8];
#pragma unroll
      for (int i = 0; i < 8; ++i) {
        const int idx4 = ((g * 8 + i) * 32 + lane) * 4;
        t[i] = *(const float4*)(K + (size_t)(kb0 + (idx4 >> 6)) * kD + h * kHS + (idx4 & 63));
      }
#pragma unroll
      for (int i = 0; i < 8; ++i) {
        const int idx4 = ((g * 8 + i) * 32 + lane) * 4;
        store4bf(&Ks[idx4 >> 6][idx4 & 63], t[i]);
      }
    }
#pragma unroll
    for (int g = 0; g < 2; ++g) {
      float4 t[8];
#pragma unroll
      for (int i = 0; i < 8; ++i) {
        const int idx4 = ((g * 8 + i) * 32 + lane) * 4;
        t[i] = *(const float4*)(Vm + (size_t)(kb0 + (idx4 >> 6)) * kD + h * kHS + (idx4 & 63));
      }
#pragma unroll
      for (int i = 0; i < 8; ++i) {
        const int idx4 = ((g * 8 + i) * 32 + lane) * 4;
        const int r = idx4 >> 6, c = idx4 & 63;
        Vt[c + 0][r] = (__bf16)t[i].x;
        Vt[c + 1][r] = (__bf16)t[i].y;
        Vt[c + 2][r] = (__bf16)t[i].z;
        Vt[c + 3][r] = (__bf16)t[i].w;
      }
    }
    __syncthreads();

    const v16bf k00 = load_frag16(&Ks[l15][khalf]);
    const v16bf k01 = load_frag16(&Ks[l15][32 + khalf]);
    const v16bf k10 = load_frag16(&Ks[16 + l15][khalf]);
    const v16bf k11 = load_frag16(&Ks[16 + l15][32 + khalf]);

    v8f s0 = zero8(), s1 = zero8();
    s0 = __builtin_amdgcn_wmma_f32_16x16x32_bf16(false, qf0, false, k00, (short)0, s0, false, false);
    s0 = __builtin_amdgcn_wmma_f32_16x16x32_bf16(false, qf1, false, k01, (short)0, s0, false, false);
    s1 = __builtin_amdgcn_wmma_f32_16x16x32_bf16(false, qf0, false, k10, (short)0, s1, false, false);
    s1 = __builtin_amdgcn_wmma_f32_16x16x32_bf16(false, qf1, false, k11, (short)0, s1, false, false);

    // online softmax: each 16-lane group owns rows {rowoff..rowoff+7} x 16 cols
#pragma unroll
    for (int r = 0; r < 8; ++r) {
      const int mg = q0 + rowoff + r;
      const float x0 = ((kb0 + l15)      <= mg) ? s0[r] * scale : -3.0e38f;
      const float x1 = ((kb0 + 16 + l15) <= mg) ? s1[r] * scale : -3.0e38f;
      float mx = fmaxf(x0, x1);
      mx = fmaxf(mx, __shfl_xor(mx, 1));
      mx = fmaxf(mx, __shfl_xor(mx, 2));
      mx = fmaxf(mx, __shfl_xor(mx, 4));
      mx = fmaxf(mx, __shfl_xor(mx, 8));
      const float mnew  = fmaxf(rmax[r], mx);
      const float alpha = __expf(rmax[r] - mnew);
      const float p0 = __expf(x0 - mnew);
      const float p1 = __expf(x1 - mnew);
      float ps = p0 + p1;
      ps += __shfl_xor(ps, 1);
      ps += __shfl_xor(ps, 2);
      ps += __shfl_xor(ps, 4);
      ps += __shfl_xor(ps, 8);
      rsum[r] = rsum[r] * alpha + ps;
      rmax[r] = mnew;
      oacc[0][r] = oacc[0][r] * alpha;
      oacc[1][r] = oacc[1][r] * alpha;
      oacc[2][r] = oacc[2][r] * alpha;
      oacc[3][r] = oacc[3][r] * alpha;
      Ps[rowoff + r][l15]      = (__bf16)p0;
      Ps[rowoff + r][16 + l15] = (__bf16)p1;
    }
    __syncthreads();

    const v16bf pf = load_frag16(&Ps[l15][khalf]);
#pragma unroll
    for (int t = 0; t < 4; ++t) {
      const v16bf vf = load_frag16(&Vt[t * 16 + l15][khalf]);
      oacc[t] = __builtin_amdgcn_wmma_f32_16x16x32_bf16(
          false, pf, false, vf, (short)0, oacc[t], false, false);
    }
    __syncthreads();
  }

#pragma unroll
  for (int t = 0; t < 4; ++t)
#pragma unroll
    for (int r = 0; r < 8; ++r) {
      const int m = q0 + rowoff + r;
      const int n = h * kHS + t * 16 + l15;
      O[(size_t)m * kD + n] = oacc[t][r] / rsum[r];
    }
}

// ---------------------------------------------------------------------------
// Host orchestration (graph-capture safe: stream-only, no alloc/sync)
// ---------------------------------------------------------------------------
extern "C" void kernel_launch(void* const* d_in, const int* in_sizes, int n_in,
                              void* d_out, int out_size, void* d_ws, size_t ws_size,
                              hipStream_t stream) {
  (void)in_sizes; (void)n_in; (void)out_size; (void)ws_size;

  const int*   idx  = (const int*)  d_in[0];
  const float* tok  = (const float*)d_in[1];
  const float* pos  = (const float*)d_in[2];
  const float* mem  = (const float*)d_in[3];
  const float* Wq   = (const float*)d_in[4];
  const float* Wk   = (const float*)d_in[5];
  const float* Wv   = (const float*)d_in[6];
  const float* Wo   = (const float*)d_in[7];
  const float* bo   = (const float*)d_in[8];
  const float* ln1g = (const float*)d_in[9];
  const float* ln1b = (const float*)d_in[10];
  const float* ln2g = (const float*)d_in[11];
  const float* ln2b = (const float*)d_in[12];
  const float* W1   = (const float*)d_in[13];
  const float* b1   = (const float*)d_in[14];
  const float* W2   = (const float*)d_in[15];
  const float* b2   = (const float*)d_in[16];
  const float* lnfg = (const float*)d_in[17];
  const float* lnfb = (const float*)d_in[18];
  const float* Wout = (const float*)d_in[19];
  float* out = (float*)d_out;

  // workspace carve-up (~110 MB of fp32); S-row buffers padded to kSP rows
  float* p   = (float*)d_ws;
  float* x_  = p; p += (size_t)kT  * kD;
  float* xa  = p; p += (size_t)kSP * kD;
  float* xn  = p; p += (size_t)kSP * kD;
  float* qb  = p; p += (size_t)kSP * kD;
  float* kb  = p; p += (size_t)kSP * kD;
  float* vb  = p; p += (size_t)kSP * kD;
  float* ob  = p; p += (size_t)kSP * kD;
  float* wpk = p; p += (size_t)kD  * kD;
  float* h1  = p; p += (size_t)kT  * kF;
  float* tb  = p; p += (size_t)kT  * kD;

  const dim3 blk256(256), blk32(32);
  const dim3 gSD(kD / BN,     kSP / BM);   // (8, 17)
  const dim3 gTF(kF / BN,     kT / BM);    // (32, 16)
  const dim3 gTD(kD / BN,     kT / BM);    // (8, 16)
  const dim3 gTV(kVocab / BN, kT / BM);    // (250, 16)
  const dim3 gAttn(kS / 16, kH);           // (132, 16)

  embed_kernel<<<kT, blk256, 0, stream>>>(idx, tok, pos, x_);

  for (int l = 0; l < kL; ++l) {
    build_xa_kernel<<<kS, blk256, 0, stream>>>(mem, x_, xa);
    layernorm_kernel<<<kS, blk256, 0, stream>>>(xa, xn, ln1g + l * kD, ln1b + l * kD);

    // Q, K, V projections (per-head weights repacked to D x D)
    pack_qkv_kernel<<<(kD * kD) / 256, blk256, 0, stream>>>(Wq + (size_t)l * kH * kD * kHS, wpk);
    gemm_bf16_kernel<<<gSD, blk256, 0, stream>>>(xn, wpk, qb, nullptr, nullptr, kSP, kD, kD, 0);
    pack_qkv_kernel<<<(kD * kD) / 256, blk256, 0, stream>>>(Wk + (size_t)l * kH * kD * kHS, wpk);
    gemm_bf16_kernel<<<gSD, blk256, 0, stream>>>(xn, wpk, kb, nullptr, nullptr, kSP, kD, kD, 0);
    pack_qkv_kernel<<<(kD * kD) / 256, blk256, 0, stream>>>(Wv + (size_t)l * kH * kD * kHS, wpk);
    gemm_bf16_kernel<<<gSD, blk256, 0, stream>>>(xn, wpk, vb, nullptr, nullptr, kSP, kD, kD, 0);

    attn_kernel<<<gAttn, blk32, 0, stream>>>(qb, kb, vb, ob);

    // xa = xa + o @ Wo + bo
    gemm_bf16_kernel<<<gSD, blk256, 0, stream>>>(ob, Wo + (size_t)l * kD * kD, xa,
                                                 bo + l * kD, xa, kSP, kD, kD, 0);

    // FFN on xt = xa[M:]
    layernorm_kernel<<<kT, blk256, 0, stream>>>(xa + (size_t)kM * kD, tb,
                                                ln2g + l * kD, ln2b + l * kD);
    gemm_bf16_kernel<<<gTF, blk256, 0, stream>>>(tb, W1 + (size_t)l * kD * kF, h1,
                                                 b1 + l * kF, nullptr, kT, kF, kD, 1);
    gemm_bf16_kernel<<<gTD, blk256, 0, stream>>>(h1, W2 + (size_t)l * kF * kD, x_,
                                                 b2 + l * kD, xa + (size_t)kM * kD,
                                                 kT, kD, kF, 0);
  }

  layernorm_kernel<<<kT, blk256, 0, stream>>>(x_, tb, lnfg, lnfb);
  gemm_bf16_kernel<<<gTV, blk256, 0, stream>>>(tb, Wout, out, nullptr, nullptr,
                                               kT, kVocab, kD, 0);
}
